// MultiHeadAttentionFusion_81295140979366
// MI455X (gfx1250) — matrix operand order repaired and checked by reading, not compiled, not run on hardware
//
#include <hip/hip_runtime.h>
#include <hip/hip_bf16.h>
#include <stdint.h>

#define E_DIM 1024
#define NHEAD 8
#define HDIM  128
#define NB    16384
#define LN_EPS 1e-5f

typedef __bf16 bf16_t;
typedef __attribute__((ext_vector_type(16))) __bf16 v16bf;
typedef __attribute__((ext_vector_type(8)))  float  v8f;

union AB32 { v16bf v; float4 q[2]; };

static __device__ __forceinline__ uint32_t f2bf(float f) {
  union { float f; uint32_t u; } c; c.f = f;
  uint32_t u = c.u;
  u += 0x7FFFu + ((u >> 16) & 1u);   // round-to-nearest-even
  return u >> 16;
}

// ---------------------------------------------------------------------------
// Stage 0: convert an f32 matrix to bf16 (weights; done once per launch)
// ---------------------------------------------------------------------------
__global__ void __launch_bounds__(256) convert_w(const float* __restrict__ w,
                                                 bf16_t* __restrict__ o, int n) {
  int i = (blockIdx.x * blockDim.x + threadIdx.x) * 4;
  if (i < n) {
    float4 f = *(const float4*)(w + i);
    uint2 p;
    p.x = f2bf(f.x) | (f2bf(f.y) << 16);
    p.y = f2bf(f.z) | (f2bf(f.w) << 16);
    *(uint2*)((unsigned short*)o + i) = p;
  }
}

// ---------------------------------------------------------------------------
// Stage 1: fused QKV projection + 2-key softmax attention.
// Block = 64 rows x one head (N-tile of 128 == head dim D).
// 8 waves arranged 2(M) x 4(N); wave tile 32x32 = 2x2 WMMA tiles.
// Produces ctx (B x E) in bf16; q/k/v never leave the block.
// ---------------------------------------------------------------------------
__global__ void __launch_bounds__(256) qkv_attn_kernel(
    const float* __restrict__ xi, const float* __restrict__ xs,
    const bf16_t* __restrict__ wq, const bf16_t* __restrict__ wk,
    const bf16_t* __restrict__ wv,
    const float* __restrict__ bq, const float* __restrict__ bk,
    const float* __restrict__ bv,
    bf16_t* __restrict__ ctx)
{
  constexpr int BM = 64, KT = 32;
  __shared__ bf16_t Ai[BM][KT];
  __shared__ bf16_t As[BM][KT];
  __shared__ float  sImg[BM];
  __shared__ float  sSig[BM];

  const int m0  = blockIdx.x * BM;       // row base
  const int n0  = blockIdx.y * HDIM;     // column base == head * 128
  const int tid  = threadIdx.x;
  const int wave = tid >> 5;
  const int lane = tid & 31;
  const int mw = wave >> 2;              // 0..1
  const int nw = wave & 3;               // 0..3
  const int lh  = lane >> 4;             // half-wave select
  const int l15 = lane & 15;

  v8f accQ[2][2]  = {}; v8f accKi[2][2] = {}; v8f accKs[2][2] = {};
  v8f accVi[2][2] = {}; v8f accVs[2][2] = {};

  for (int kb = 0; kb < E_DIM; kb += KT) {
    __syncthreads();
    { // stage X tiles (f32 -> bf16) into LDS
      const int r = tid >> 3;
      const int c = (tid & 7) << 2;
      #pragma unroll
      for (int rr = 0; rr < BM; rr += 32) {
        float4 fi = *(const float4*)(xi + (size_t)(m0 + r + rr) * E_DIM + kb + c);
        float4 fs = *(const float4*)(xs + (size_t)(m0 + r + rr) * E_DIM + kb + c);
        uint2 pi, ps;
        pi.x = f2bf(fi.x) | (f2bf(fi.y) << 16);
        pi.y = f2bf(fi.z) | (f2bf(fi.w) << 16);
        ps.x = f2bf(fs.x) | (f2bf(fs.y) << 16);
        ps.y = f2bf(fs.z) | (f2bf(fs.w) << 16);
        *(uint2*)&Ai[r + rr][c] = pi;
        *(uint2*)&As[r + rr][c] = ps;
      }
    }
    __syncthreads();

    // A fragments: 16-bit A layout -> lane<16: K{0..7,16..23}, lane>=16: +8
    v16bf aI[2], aS[2];
    #pragma unroll
    for (int mi = 0; mi < 2; ++mi) {
      const int row = mw * 32 + mi * 16 + l15;
      const int kc  = lh ? 8 : 0;
      AB32 t;
      t.q[0] = *(const float4*)&Ai[row][kc];
      t.q[1] = *(const float4*)&Ai[row][kc + 16];
      aI[mi] = t.v;
      t.q[0] = *(const float4*)&As[row][kc];
      t.q[1] = *(const float4*)&As[row][kc + 16];
      aS[mi] = t.v;
    }

    // B fragments straight from L2-resident bf16 weights:
    // B[k][n] = W[n][k]; lane n holds K{0..7,16..23} (+8 for upper half) of row n.
    #pragma unroll
    for (int ni = 0; ni < 2; ++ni) {
      const int n = n0 + nw * 32 + ni * 16 + l15;
      const size_t off = (size_t)n * E_DIM + kb + (lh ? 8 : 0);
      AB32 tq, tk, tv;
      tq.q[0] = *(const float4*)(wq + off); tq.q[1] = *(const float4*)(wq + off + 16);
      tk.q[0] = *(const float4*)(wk + off); tk.q[1] = *(const float4*)(wk + off + 16);
      tv.q[0] = *(const float4*)(wv + off); tv.q[1] = *(const float4*)(wv + off + 16);
      #pragma unroll
      for (int mi = 0; mi < 2; ++mi) {
        accQ[mi][ni]  = __builtin_amdgcn_wmma_f32_16x16x32_bf16(false, aI[mi], false, tq.v, (short)0, accQ[mi][ni],  false, false);
        accKi[mi][ni] = __builtin_amdgcn_wmma_f32_16x16x32_bf16(false, aI[mi], false, tk.v, (short)0, accKi[mi][ni], false, false);
        accKs[mi][ni] = __builtin_amdgcn_wmma_f32_16x16x32_bf16(false, aS[mi], false, tk.v, (short)0, accKs[mi][ni], false, false);
        accVi[mi][ni] = __builtin_amdgcn_wmma_f32_16x16x32_bf16(false, aI[mi], false, tv.v, (short)0, accVi[mi][ni], false, false);
        accVs[mi][ni] = __builtin_amdgcn_wmma_f32_16x16x32_bf16(false, aS[mi], false, tv.v, (short)0, accVs[mi][ni], false, false);
      }
    }
  }

  // ---- attention epilogue: per-row score dots over this head's 128 dims ----
  if (tid < BM) { sImg[tid] = 0.f; sSig[tid] = 0.f; }
  __syncthreads();

  const int mBase = mw * 32;
  #pragma unroll
  for (int mi = 0; mi < 2; ++mi) {
    #pragma unroll
    for (int r = 0; r < 8; ++r) {
      const int m = mBase + mi * 16 + r + 8 * lh;
      float pi = 0.f, ps = 0.f;
      #pragma unroll
      for (int ni = 0; ni < 2; ++ni) {
        const int n = n0 + nw * 32 + ni * 16 + l15;
        const float qv = accQ[mi][ni][r]  + bq[n];
        const float ki = accKi[mi][ni][r] + bk[n];
        const float ks = accKs[mi][ni][r] + bk[n];
        pi += qv * ki;
        ps += qv * ks;
      }
      atomicAdd(&sImg[m], pi);
      atomicAdd(&sSig[m], ps);
    }
  }
  __syncthreads();

  const float scale = 0.08838834764831845f;   // 1/sqrt(128)
  unsigned short* ctx_us = (unsigned short*)ctx;
  #pragma unroll
  for (int mi = 0; mi < 2; ++mi) {
    #pragma unroll
    for (int r = 0; r < 8; ++r) {
      const int m = mBase + mi * 16 + r + 8 * lh;
      const float s0 = sImg[m] * scale;
      const float s1 = sSig[m] * scale;
      const float mx = fmaxf(s0, s1);
      const float e0 = __expf(s0 - mx);
      const float e1 = __expf(s1 - mx);
      const float inv = 1.f / (e0 + e1);
      const float a0 = e0 * inv, a1 = e1 * inv;
      #pragma unroll
      for (int ni = 0; ni < 2; ++ni) {
        const int n = n0 + nw * 32 + ni * 16 + l15;
        const float vi = accVi[mi][ni][r] + bv[n];
        const float vs = accVs[mi][ni][r] + bv[n];
        const float c  = a0 * vi + a1 * vs;
        ctx_us[(size_t)(m0 + m) * E_DIM + n] = (unsigned short)f2bf(c);
      }
    }
  }
}

// ---------------------------------------------------------------------------
// Stage 2: out = LayerNorm(ctx @ Wo^T + bo + x_img) * gamma + beta
// Block = 32 rows x full E (so LN rows complete inside one block).
// 8 waves each own a 128-wide N slice; wave tile 32x128 = 2x8 WMMA tiles.
// ---------------------------------------------------------------------------
__global__ void __launch_bounds__(256) out_ln_kernel(
    const bf16_t* __restrict__ ctx, const bf16_t* __restrict__ wo,
    const float* __restrict__ bo, const float* __restrict__ xi,
    const float* __restrict__ gamma, const float* __restrict__ beta,
    float* __restrict__ out)
{
  constexpr int BM = 32, KT = 32;
  __shared__ bf16_t Ac[BM][KT];
  __shared__ float  sSum[BM];
  __shared__ float  sSq[BM];

  const int m0  = blockIdx.x * BM;
  const int tid  = threadIdx.x;
  const int wave = tid >> 5;
  const int lane = tid & 31;
  const int lh  = lane >> 4;
  const int l15 = lane & 15;
  const int nwBase = wave * 128;

  v8f acc[2][8] = {};

  for (int kb = 0; kb < E_DIM; kb += KT) {
    __syncthreads();
    { // stage ctx tile 32x32 bf16 (already bf16 in workspace)
      const int r = tid >> 3;
      const int c = (tid & 7) << 2;
      *(uint2*)&Ac[r][c] = *(const uint2*)(ctx + (size_t)(m0 + r) * E_DIM + kb + c);
    }
    __syncthreads();

    v16bf a[2];
    #pragma unroll
    for (int mi = 0; mi < 2; ++mi) {
      const int row = mi * 16 + l15;
      const int kc  = lh ? 8 : 0;
      AB32 t;
      t.q[0] = *(const float4*)&Ac[row][kc];
      t.q[1] = *(const float4*)&Ac[row][kc + 16];
      a[mi] = t.v;
    }
    #pragma unroll
    for (int ni = 0; ni < 8; ++ni) {
      const int n = nwBase + ni * 16 + l15;
      const size_t off = (size_t)n * E_DIM + kb + (lh ? 8 : 0);
      AB32 t;
      t.q[0] = *(const float4*)(wo + off);
      t.q[1] = *(const float4*)(wo + off + 16);
      #pragma unroll
      for (int mi = 0; mi < 2; ++mi)
        acc[mi][ni] = __builtin_amdgcn_wmma_f32_16x16x32_bf16(false, a[mi], false, t.v, (short)0, acc[mi][ni], false, false);
    }
  }

  // ---- epilogue: bias + residual, then LayerNorm across E via LDS ----
  if (tid < BM) { sSum[tid] = 0.f; sSq[tid] = 0.f; }
  __syncthreads();

  #pragma unroll
  for (int mi = 0; mi < 2; ++mi) {
    #pragma unroll
    for (int r = 0; r < 8; ++r) {
      const int m = mi * 16 + r + 8 * lh;
      float s = 0.f, ss = 0.f;
      #pragma unroll
      for (int ni = 0; ni < 8; ++ni) {
        const int n = nwBase + ni * 16 + l15;
        float v = acc[mi][ni][r] + bo[n] + xi[(size_t)(m0 + m) * E_DIM + n];
        acc[mi][ni][r] = v;          // keep fused value in the accumulator
        s += v; ss += v * v;
      }
      atomicAdd(&sSum[m], s);
      atomicAdd(&sSq[m], ss);
    }
  }
  __syncthreads();

  const float invE = 1.f / (float)E_DIM;
  #pragma unroll
  for (int mi = 0; mi < 2; ++mi) {
    #pragma unroll
    for (int r = 0; r < 8; ++r) {
      const int m  = mi * 16 + r + 8 * lh;
      const float mu  = sSum[m] * invE;
      const float var = sSq[m] * invE - mu * mu;
      const float inv = rsqrtf(var + LN_EPS);
      #pragma unroll
      for (int ni = 0; ni < 8; ++ni) {
        const int n = nwBase + ni * 16 + l15;
        const float v = (acc[mi][ni][r] - mu) * inv;
        out[(size_t)(m0 + m) * E_DIM + n] = v * gamma[n] + beta[n];
      }
    }
  }
}

// ---------------------------------------------------------------------------
extern "C" void kernel_launch(void* const* d_in, const int* in_sizes, int n_in,
                              void* d_out, int out_size, void* d_ws, size_t ws_size,
                              hipStream_t stream) {
  const float* xi    = (const float*)d_in[0];
  const float* xs    = (const float*)d_in[1];
  const float* Wq    = (const float*)d_in[2];
  const float* Wk    = (const float*)d_in[3];
  const float* Wv    = (const float*)d_in[4];
  const float* bq    = (const float*)d_in[5];
  const float* bk    = (const float*)d_in[6];
  const float* bv    = (const float*)d_in[7];
  const float* Wo    = (const float*)d_in[8];
  const float* bo    = (const float*)d_in[9];
  const float* gamma = (const float*)d_in[10];
  const float* beta  = (const float*)d_in[11];
  float* out = (float*)d_out;

  const size_t EE = (size_t)E_DIM * E_DIM;
  bf16_t* wq_b = (bf16_t*)d_ws;
  bf16_t* wk_b = wq_b + EE;
  bf16_t* wv_b = wk_b + EE;
  bf16_t* wo_b = wv_b + EE;
  bf16_t* ctxb = wo_b + EE;              // NB * E_DIM bf16

  const int cvt_blocks = (int)(EE / (256 * 4));
  convert_w<<<cvt_blocks, 256, 0, stream>>>(Wq, wq_b, (int)EE);
  convert_w<<<cvt_blocks, 256, 0, stream>>>(Wk, wk_b, (int)EE);
  convert_w<<<cvt_blocks, 256, 0, stream>>>(Wv, wv_b, (int)EE);
  convert_w<<<cvt_blocks, 256, 0, stream>>>(Wo, wo_b, (int)EE);

  dim3 g1(NB / 64, NHEAD);
  qkv_attn_kernel<<<g1, 256, 0, stream>>>(xi, xs, wq_b, wk_b, wv_b,
                                          bq, bk, bv, ctxb);

  out_ln_kernel<<<NB / 32, 256, 0, stream>>>(ctxb, wo_b, bo, xi, gamma, beta, out);
}
